// _Attn2048_10849087390342
// MI455X (gfx1250) — compile-verified, matrix operand order checked by reading, not run on hardware
//
#include <hip/hip_runtime.h>
#include <hip/hip_bf16.h>

// Problem constants: B=2, L=4096, D=512, H=8, DK=64
#define BB 2
#define LL 4096
#define DD 512
#define HH 8
#define DKK 64

typedef __bf16 bf16;
typedef __attribute__((ext_vector_type(16))) __bf16 v16bf;
typedef __attribute__((ext_vector_type(8)))  __bf16 v8bf;
typedef __attribute__((ext_vector_type(8)))  float  v8f;

union BF16x16 { v16bf v; v8bf h[2]; };

__device__ __forceinline__ v16bf ld_frag(const bf16* p0, const bf16* p1) {
    BF16x16 f;
    f.h[0] = *(const v8bf*)p0;
    f.h[1] = *(const v8bf*)p1;
    return f.v;
}

__device__ __forceinline__ v8f wmma_bf(v16bf a, v16bf b, v8f c) {
    return __builtin_amdgcn_wmma_f32_16x16x32_bf16(false, a, false, b, (short)0, c, false, false);
}

// ---------------- f32 -> bf16 convert (8 elems / thread, b128 stores) ----------------
__global__ void cvt_kernel(const float* __restrict__ in, bf16* __restrict__ out, int n8) {
    int i = blockIdx.x * blockDim.x + threadIdx.x;
    if (i < n8) {
        const float* p = in + (size_t)i * 8;
        v8bf v;
#pragma unroll
        for (int j = 0; j < 8; ++j) v[j] = (bf16)p[j];
        *(v8bf*)(out + (size_t)i * 8) = v;
    }
}

// ---------------- QKV projection GEMM (32x64 per wave, pipelined) ----------------
// X[8192,512] bf16 @ W[1536,512]^T -> Q[B,H,L,64], K[B,H,L,64], Vt[B,H,64,L]
__global__ void __launch_bounds__(256) qkv_gemm_kernel(
    const bf16* __restrict__ X, const bf16* __restrict__ W,
    bf16* __restrict__ Qp, bf16* __restrict__ Kp, bf16* __restrict__ Vt)
{
    const int lane = threadIdx.x & 31;
    const int half = lane >> 4, c16 = lane & 15;
    const int wave = (blockIdx.x * blockDim.x + threadIdx.x) >> 5;
    const int NT = (3 * DD) / 64;               // 24 column groups
    const int nt = wave % NT, mt = wave / NT;   // mt in [0,256)
    const int m0 = mt * 32, n0 = nt * 64;

    const bf16* xrow0 = X + (size_t)(m0 + c16) * DD;
    const bf16* xrow1 = X + (size_t)(m0 + 16 + c16) * DD;
    const bf16* wrow0 = W + (size_t)(n0 + 0 * 16 + c16) * DD;
    const bf16* wrow1 = W + (size_t)(n0 + 1 * 16 + c16) * DD;
    const bf16* wrow2 = W + (size_t)(n0 + 2 * 16 + c16) * DD;
    const bf16* wrow3 = W + (size_t)(n0 + 3 * 16 + c16) * DD;

    v8f acc[2][4] = {};

    auto LD = [&](int k0, v16bf* a, v16bf* b) {
        a[0] = ld_frag(xrow0 + k0 + half * 8, xrow0 + k0 + 16 + half * 8);
        a[1] = ld_frag(xrow1 + k0 + half * 8, xrow1 + k0 + 16 + half * 8);
        b[0] = ld_frag(wrow0 + k0 + half * 8, wrow0 + k0 + 16 + half * 8);
        b[1] = ld_frag(wrow1 + k0 + half * 8, wrow1 + k0 + 16 + half * 8);
        b[2] = ld_frag(wrow2 + k0 + half * 8, wrow2 + k0 + 16 + half * 8);
        b[3] = ld_frag(wrow3 + k0 + half * 8, wrow3 + k0 + 16 + half * 8);
    };
    auto MM = [&](v16bf* a, v16bf* b) {
#pragma unroll
        for (int i = 0; i < 2; ++i)
#pragma unroll
            for (int j = 0; j < 4; ++j)
                acc[i][j] = wmma_bf(a[i], b[j], acc[i][j]);
    };

    v16bf aA[2], bA[4], aB[2], bB[4];
    LD(0, aA, bA);
#pragma unroll 1
    for (int k0 = 0; k0 < DD; k0 += 64) {
        LD(k0 + 32, aB, bB);
        MM(aA, bA);
        if (k0 + 64 < DD) LD(k0 + 64, aA, bA);
        MM(aB, bB);
    }

    // scatter
#pragma unroll
    for (int rs = 0; rs < 2; ++rs) {
#pragma unroll
        for (int j = 0; j < 4; ++j) {
            const int gcol = n0 + j * 16 + c16;
            const int which = gcol >> 9;           // 0=Q 1=K 2=V
            const int d = gcol & (DD - 1);
            const int h = d >> 6, dk = d & 63;
            const int mbase = m0 + rs * 16 + half * 8;   // aligned to 8
            const int b = mbase >> 12;
            const int l0 = mbase & (LL - 1);
            const size_t bh = (size_t)(b * HH + h);
            if (which == 2) {
                v8bf pv;
#pragma unroll
                for (int r = 0; r < 8; ++r) pv[r] = (bf16)acc[rs][j][r];
                *(v8bf*)(Vt + (bh * DKK + dk) * LL + l0) = pv;   // contiguous in L
            } else {
                bf16* dst = (which == 0) ? Qp : Kp;
#pragma unroll
                for (int r = 0; r < 8; ++r)
                    dst[(bh * LL + (l0 + r)) * DKK + dk] = (bf16)acc[rs][j][r];
            }
        }
    }
}

// ---------------- Flash attention (ping-pong K, V loads overlap softmax) -------------
__global__ void __launch_bounds__(256) attn_kernel(
    const bf16* __restrict__ Qp, const bf16* __restrict__ Kp,
    const bf16* __restrict__ Vt, bf16* __restrict__ Ctx)
{
    __shared__ bf16 pstage[8 * 16 * 32];
    const int lane = threadIdx.x & 31;
    const int wib  = threadIdx.x >> 5;
    const int half = lane >> 4, c16 = lane & 15;
    const int wave = blockIdx.x * 8 + wib;       // 0..4095
    const int qt = wave & 255;
    const int bh = wave >> 8;                    // 0..15
    const int q0 = qt * 16;

    const bf16* Qb = Qp + (size_t)bh * LL * DKK;
    const bf16* Kb = Kp + (size_t)bh * LL * DKK;
    const bf16* Vb = Vt + (size_t)bh * DKK * LL;

    // load Q fragments and fold in SCALE = 0.125 (power of two -> exact in bf16)
    const bf16* qrow = Qb + (size_t)(q0 + c16) * DKK;
    v16bf aq0 = ld_frag(qrow + half * 8,      qrow + 16 + half * 8);
    v16bf aq1 = ld_frag(qrow + 32 + half * 8, qrow + 48 + half * 8);
#pragma unroll
    for (int i = 0; i < 16; ++i) {
        aq0[i] = (bf16)((float)aq0[i] * 0.125f);
        aq1[i] = (bf16)((float)aq1[i] * 0.125f);
    }

    v8f acc[4] = {};
    float mrow[8], lrow[8];
#pragma unroll
    for (int r = 0; r < 8; ++r) { mrow[r] = -INFINITY; lrow[r] = 0.0f; }

    bf16* pbuf = pstage + wib * (16 * 32);

    auto LDK = [&](int kv, v16bf* bk) {
        const bf16* k0r = Kb + (size_t)(kv + c16) * DKK;
        const bf16* k1r = Kb + (size_t)(kv + 16 + c16) * DKK;
        bk[0] = ld_frag(k0r + half * 8,      k0r + 16 + half * 8);
        bk[1] = ld_frag(k0r + 32 + half * 8, k0r + 48 + half * 8);
        bk[2] = ld_frag(k1r + half * 8,      k1r + 16 + half * 8);
        bk[3] = ld_frag(k1r + 32 + half * 8, k1r + 48 + half * 8);
    };

    auto STEP = [&](int kv, v16bf* bk, v16bf* bknext, bool loadnext) {
        // S = Q @ K^T (16x32), f32 accum
        v8f s0 = {}, s1 = {};
        s0 = wmma_bf(aq0, bk[0], s0); s0 = wmma_bf(aq1, bk[1], s0);
        s1 = wmma_bf(aq0, bk[2], s1); s1 = wmma_bf(aq1, bk[3], s1);

        // issue V loads for this step + K loads for next step BEFORE softmax,
        // so the VALU/trans-heavy softmax overlaps the memory latency
        v16bf bv[4];
#pragma unroll
        for (int j = 0; j < 4; ++j) {
            const bf16* vr = Vb + (size_t)(j * 16 + c16) * LL + kv;
            bv[j] = ld_frag(vr + half * 8, vr + 16 + half * 8);
        }
        if (loadnext) LDK(kv + 32, bknext);

        // online softmax
#pragma unroll
        for (int r = 0; r < 8; ++r) {
            float x0 = s0[r], x1 = s1[r];
            float mx = fmaxf(x0, x1);
#pragma unroll
            for (int o = 1; o < 16; o <<= 1) mx = fmaxf(mx, __shfl_xor(mx, o, 32));
            const float mn = fmaxf(mrow[r], mx);
            const float alpha = __expf(mrow[r] - mn);
            const float p0 = __expf(x0 - mn), p1 = __expf(x1 - mn);
            float rs = p0 + p1;
#pragma unroll
            for (int o = 1; o < 16; o <<= 1) rs += __shfl_xor(rs, o, 32);
            lrow[r] = lrow[r] * alpha + rs;
            mrow[r] = mn;
#pragma unroll
            for (int j = 0; j < 4; ++j) acc[j][r] *= alpha;
            pbuf[(half * 8 + r) * 32 + c16]      = (bf16)p0;
            pbuf[(half * 8 + r) * 32 + 16 + c16] = (bf16)p1;
        }
        asm volatile("s_wait_dscnt 0" ::: "memory");

        // P (16x32) as A-fragment from LDS; O += P @ V
        const bf16* prow = pbuf + c16 * 32;
        v16bf ap = ld_frag(prow + half * 8, prow + 16 + half * 8);
#pragma unroll
        for (int j = 0; j < 4; ++j) acc[j] = wmma_bf(ap, bv[j], acc[j]);
    };

    v16bf bkA[4], bkB[4];
    LDK(0, bkA);
#pragma unroll 1
    for (int kv = 0; kv < LL; kv += 64) {
        STEP(kv,      bkA, bkB, true);
        STEP(kv + 32, bkB, bkA, kv + 64 < LL);
    }

    // epilogue: O /= l, write ctx_bf [B*L, 512]
    const int b = bh >> 3, h = bh & 7;
#pragma unroll
    for (int j = 0; j < 4; ++j) {
        const int dk = j * 16 + c16;
#pragma unroll
        for (int r = 0; r < 8; ++r) {
            const int l = q0 + half * 8 + r;
            const float val = acc[j][r] / lrow[r];
            Ctx[((size_t)(b * LL + l)) * DD + h * DKK + dk] = (bf16)val;
        }
    }
}

// ---------------- Output projection GEMM (32x64 per wave, pipelined) ----------------
// Ctx[8192,512] bf16 @ Wo[512,512]^T -> out f32 [8192,512]
__global__ void __launch_bounds__(256) out_gemm_kernel(
    const bf16* __restrict__ C, const bf16* __restrict__ W, float* __restrict__ out)
{
    const int lane = threadIdx.x & 31;
    const int half = lane >> 4, c16 = lane & 15;
    const int wave = (blockIdx.x * blockDim.x + threadIdx.x) >> 5;
    const int NT = DD / 64;                      // 8
    const int nt = wave % NT, mt = wave / NT;    // mt in [0,256)
    const int m0 = mt * 32, n0 = nt * 64;

    const bf16* crow0 = C + (size_t)(m0 + c16) * DD;
    const bf16* crow1 = C + (size_t)(m0 + 16 + c16) * DD;
    const bf16* wrow0 = W + (size_t)(n0 + 0 * 16 + c16) * DD;
    const bf16* wrow1 = W + (size_t)(n0 + 1 * 16 + c16) * DD;
    const bf16* wrow2 = W + (size_t)(n0 + 2 * 16 + c16) * DD;
    const bf16* wrow3 = W + (size_t)(n0 + 3 * 16 + c16) * DD;

    v8f acc[2][4] = {};

    auto LD = [&](int k0, v16bf* a, v16bf* b) {
        a[0] = ld_frag(crow0 + k0 + half * 8, crow0 + k0 + 16 + half * 8);
        a[1] = ld_frag(crow1 + k0 + half * 8, crow1 + k0 + 16 + half * 8);
        b[0] = ld_frag(wrow0 + k0 + half * 8, wrow0 + k0 + 16 + half * 8);
        b[1] = ld_frag(wrow1 + k0 + half * 8, wrow1 + k0 + 16 + half * 8);
        b[2] = ld_frag(wrow2 + k0 + half * 8, wrow2 + k0 + 16 + half * 8);
        b[3] = ld_frag(wrow3 + k0 + half * 8, wrow3 + k0 + 16 + half * 8);
    };
    auto MM = [&](v16bf* a, v16bf* b) {
#pragma unroll
        for (int i = 0; i < 2; ++i)
#pragma unroll
            for (int j = 0; j < 4; ++j)
                acc[i][j] = wmma_bf(a[i], b[j], acc[i][j]);
    };

    v16bf aA[2], bA[4], aB[2], bB[4];
    LD(0, aA, bA);
#pragma unroll 1
    for (int k0 = 0; k0 < DD; k0 += 64) {
        LD(k0 + 32, aB, bB);
        MM(aA, bA);
        if (k0 + 64 < DD) LD(k0 + 64, aA, bA);
        MM(aB, bB);
    }

#pragma unroll
    for (int rs = 0; rs < 2; ++rs)
#pragma unroll
        for (int j = 0; j < 4; ++j)
#pragma unroll
            for (int r = 0; r < 8; ++r) {
                const int m = m0 + rs * 16 + half * 8 + r;
                out[(size_t)m * DD + n0 + j * 16 + c16] = acc[rs][j][r];
            }
}

extern "C" void kernel_launch(void* const* d_in, const int* in_sizes, int n_in,
                              void* d_out, int out_size, void* d_ws, size_t ws_size,
                              hipStream_t stream) {
    const float* x     = (const float*)d_in[0];   // [2,4096,512]
    const float* w_qkv = (const float*)d_in[1];   // [1536,512]
    const float* w_o   = (const float*)d_in[2];   // [512,512]
    float* out = (float*)d_out;                   // [2,4096,512] f32

    size_t off = 0;
    auto alloc = [&](size_t bytes) -> void* {
        void* p = (char*)d_ws + off;
        off += (bytes + 255) & ~(size_t)255;
        return p;
    };
    const size_t nX  = (size_t)BB * LL * DD;        // 4194304
    const size_t nWq = (size_t)3 * DD * DD;         // 786432
    const size_t nWo = (size_t)DD * DD;             // 262144
    const size_t nQ  = (size_t)BB * HH * LL * DKK;  // 4194304

    bf16* x_bf    = (bf16*)alloc(nX  * sizeof(bf16));
    bf16* wqkv_bf = (bf16*)alloc(nWq * sizeof(bf16));
    bf16* wo_bf   = (bf16*)alloc(nWo * sizeof(bf16));
    bf16* Qp      = (bf16*)alloc(nQ  * sizeof(bf16));
    bf16* Kp      = (bf16*)alloc(nQ  * sizeof(bf16));
    bf16* Vt      = (bf16*)alloc(nQ  * sizeof(bf16));
    bf16* ctx_bf  = (bf16*)alloc(nX  * sizeof(bf16));

    cvt_kernel<<<(int)((nX / 8 + 255) / 256), 256, 0, stream>>>(x,     x_bf,    (int)(nX / 8));
    cvt_kernel<<<(int)((nWq / 8 + 255) / 256), 256, 0, stream>>>(w_qkv, wqkv_bf, (int)(nWq / 8));
    cvt_kernel<<<(int)((nWo / 8 + 255) / 256), 256, 0, stream>>>(w_o,   wo_bf,   (int)(nWo / 8));

    // QKV GEMM: 256 row tiles (32 rows) x 24 col groups = 6144 waves -> 768 blocks
    qkv_gemm_kernel<<<768, 256, 0, stream>>>(x_bf, wqkv_bf, Qp, Kp, Vt);

    // Attention: B*H*(L/16) = 4096 waves -> 512 blocks
    attn_kernel<<<512, 256, 0, stream>>>(Qp, Kp, Vt, ctx_bf);

    // Output GEMM: 256 row tiles x 8 col groups = 2048 waves -> 256 blocks
    out_gemm_kernel<<<256, 256, 0, stream>>>(ctx_bf, wo_bf, out);
}
